// CausalSelfAttention_41961830482330
// MI455X (gfx1250) — compile-verified, matrix operand order checked by reading, not run on hardware
//
#include <hip/hip_runtime.h>
#include <hip/hip_bf16.h>
#include <math.h>

// ---------------------------------------------------------------------------
// RWKV7-style block: qkv GEMM -> groupnorm+erf -> sequential scan -> proj GEMM
// MI455X (gfx1250, wave32). GEMMs: v_wmma_f32_16x16x32_bf16 with the B tile
// staged into LDS by the Tensor Data Mover (tensor_load_to_lds, TENSORcnt).
// ---------------------------------------------------------------------------

typedef __attribute__((ext_vector_type(16))) __bf16 v16bf;
typedef __attribute__((ext_vector_type(8)))  float  v8f;
typedef __attribute__((ext_vector_type(4)))  float  v4f;
typedef __attribute__((ext_vector_type(4)))  __bf16 v4bf;
typedef __attribute__((ext_vector_type(4)))  unsigned int uint4v;
typedef __attribute__((ext_vector_type(8)))  unsigned int uint8v;

#define B_DIM 4
#define T_DIM 2048
#define C_DIM 1024
#define NH_DIM 16
#define HS_DIM 64

// ---------------------------------------------------------------------------
// fp32 -> bf16 conversion, 4 elements per thread (b128 load, b64 store)
// ---------------------------------------------------------------------------
__global__ void f32_to_bf16_kernel(const float* __restrict__ in,
                                   __bf16* __restrict__ out, int n4) {
  int i = blockIdx.x * blockDim.x + threadIdx.x;
  if (i < n4) {
    v4f v = reinterpret_cast<const v4f*>(in)[i];
    reinterpret_cast<v4bf*>(out)[i] = __builtin_convertvector(v, v4bf);
  }
}

// ---------------------------------------------------------------------------
// TDM issue: 2-D tile load Global -> LDS via D# descriptor groups 0/1 in SGPRs
// (CDNA5 ISA). 2-group form => tensors up to 2D.
//   g0: count=1 | lds_addr | global_addr[56:0] | type=2
//   g1: data_size=2B, tensor_dim0=K, tensor_dim1=N_rows, tile=32x64, stride=K
// ---------------------------------------------------------------------------
__device__ __forceinline__ void tdm_issue(uint4v g0, uint8v g1) {
  asm volatile("tensor_load_to_lds %0, %1" :: "s"(g0), "s"(g1) : "memory");
}

__device__ __forceinline__ uint4v tdm_make_g0(const __bf16* g_tile,
                                              unsigned lds_byte_off) {
  unsigned long long ga = (unsigned long long)(const void*)g_tile;
  uint4v g0;
  g0[0] = 1u;                                            // count=1, user mode
  g0[1] = lds_byte_off;                                  // lds_addr (bytes)
  g0[2] = (unsigned)ga;                                  // global_addr[31:0]
  g0[3] = (unsigned)((ga >> 32) & 0x01FFFFFFu) | (2u << 30);  // [56:32] | type=2
  return g0;
}

__device__ __forceinline__ uint8v tdm_make_g1(int K, int Nrows) {
  uint8v g1;
  g1[0] = 0x00010000u;                                   // data_size=1 (2 bytes)
  g1[1] = ((unsigned)K & 0xFFFFu) << 16;                 // tensor_dim0 lo16
  g1[2] = (((unsigned)K >> 16) & 0xFFFFu) | (((unsigned)Nrows & 0xFFFFu) << 16);
  g1[3] = (((unsigned)Nrows >> 16) & 0xFFFFu) | (32u << 16);  // tile_dim0 = 32
  g1[4] = 64u;                                           // tile_dim1=64, dim2=0
  g1[5] = (unsigned)K;                                   // dim0_stride lo32
  g1[6] = 0u;                                            // stride hi, dim1_stride
  g1[7] = 0u;
  return g1;
}

// ---------------------------------------------------------------------------
// Fragment loaders for v_wmma_f32_16x16x32_bf16 (ISA 7.12.2 layouts).
//   lanes 0-15 : row = lane,     K banks {0..7, 16..23}
//   lanes 16-31: row = lane-16,  K banks {8..15, 24..31}
// Global (row-major, ld elements): two b128 per lane at k_base+8*half (+32B).
// ---------------------------------------------------------------------------
__device__ __forceinline__ v16bf load_frag_global(const __bf16* __restrict__ base,
                                                  int ld, int rc_base, int k_base,
                                                  int lane) {
  const int rc   = rc_base + (lane & 15);
  const int half = lane >> 4;
  const __bf16* p = base + (size_t)rc * ld + k_base + 8 * half;
  union { v16bf v; unsigned int u[8]; } f;
#pragma unroll
  for (int r = 0; r < 8; ++r) {
    const int k = 2 * r + (r >= 4 ? 8 : 0);
    f.u[r] = *reinterpret_cast<const unsigned int*>(p + k);
  }
  return f.v;
}

// LDS tile: 64 rows x 32 bf16, rows contiguous (64B) as written by the TDM.
__device__ __forceinline__ v16bf load_frag_lds(const __bf16* smem, int row_base,
                                               int lane) {
  const int row  = row_base + (lane & 15);
  const int half = lane >> 4;
  const __bf16* p = smem + row * 32 + half * 8;
  union { v16bf v; uint4v q[2]; } f;
  f.q[0] = *reinterpret_cast<const uint4v*>(p);
  f.q[1] = *reinterpret_cast<const uint4v*>(p + 16);
  return f.v;
}

// ---------------------------------------------------------------------------
// GEMM: C[M,N](f32) = A[M,K](bf16) * W[N,K](bf16)^T  (+ fp32 residual if RES)
// block = 128 threads = 4 waves stacked on M; wave tile 32x64 (2x4 accums,
// 8 v_wmma per K-step). B tile (64x32) TDM-staged into LDS, double-buffered:
// wave 0 issues tensor_load_to_lds for step k+1, waits TENSORcnt for step k,
// then a workgroup barrier publishes the tile. A fragments load from global.
// grid = (M/128, N/64); M,N,K multiples of tile sizes here.
// ---------------------------------------------------------------------------
template <bool RES>
__global__ void gemm_bf16_wmma_kernel(const __bf16* __restrict__ A,
                                      const __bf16* __restrict__ W,
                                      float* __restrict__ C,
                                      const float* __restrict__ resid,
                                      int N, int K) {
  __shared__ __bf16 Bsm[2][64 * 32];  // 2 x 4KB double buffer

  const int lane = threadIdx.x & 31;
  const int wave = threadIdx.x >> 5;
  const int m0 = blockIdx.x * 128 + wave * 32;
  const int n0 = blockIdx.y * 64;

  const uint8v g1 = tdm_make_g1(K, N);
  const unsigned lds_off[2] = {
      (unsigned)(unsigned long long)(size_t)&Bsm[0][0],   // LDS byte offset =
      (unsigned)(unsigned long long)(size_t)&Bsm[1][0] }; // flat addr [31:0]

  // prologue: stage first B tile
  if (threadIdx.x < 32)
    tdm_issue(tdm_make_g0(W + (size_t)n0 * K, lds_off[0]), g1);

  v8f acc[2][4] = {};
  int buf = 0;

  for (int k0 = 0; k0 < K; k0 += 32) {
    if (threadIdx.x < 32) {
      if (k0 + 32 < K) {  // issue next tile, wait for current (in-order TDM)
        tdm_issue(tdm_make_g0(W + (size_t)n0 * K + (k0 + 32), lds_off[buf ^ 1]), g1);
        __builtin_amdgcn_s_wait_tensorcnt(1);
      } else {
        __builtin_amdgcn_s_wait_tensorcnt(0);
      }
    }
    __syncthreads();  // B tile [buf] visible to all waves

    if (k0 + 32 < K)  // prefetch next A tile (global_prefetch_b8)
      __builtin_prefetch(A + (size_t)(m0 + (lane & 15)) * K + k0 + 32, 0, 1);

    v16bf af[2], bfr[4];
#pragma unroll
    for (int i = 0; i < 2; ++i)
      af[i] = load_frag_global(A, K, m0 + 16 * i, k0, lane);
#pragma unroll
    for (int j = 0; j < 4; ++j)
      bfr[j] = load_frag_lds(&Bsm[buf][0], 16 * j, lane);
#pragma unroll
    for (int i = 0; i < 2; ++i)
#pragma unroll
      for (int j = 0; j < 4; ++j)
        acc[i][j] = __builtin_amdgcn_wmma_f32_16x16x32_bf16(
            /*neg_a=*/false, af[i], /*neg_b=*/false, bfr[j],
            /*c_mod=*/(short)0, acc[i][j], /*reuse_a=*/false, /*reuse_b=*/false);

    __syncthreads();  // all waves done with [buf] before it is refilled
    buf ^= 1;
  }

  // C/D layout: lane&15 = N, VGPR r -> M = r + 8*(lane>>4)
  const int half = lane >> 4;
  const int nl = lane & 15;
#pragma unroll
  for (int i = 0; i < 2; ++i)
#pragma unroll
    for (int j = 0; j < 4; ++j)
#pragma unroll
      for (int r = 0; r < 8; ++r) {
        const int m = m0 + i * 16 + r + half * 8;
        const int n = n0 + j * 16 + nl;
        const size_t idx = (size_t)m * N + n;
        float v = acc[i][j][r];
        if (RES) v += resid[idx];
        C[idx] = v;
      }
}

// ---------------------------------------------------------------------------
// Per-(b,t,slot,h) normalization over HS=64 with unbiased std (ddof=1),
// then erf on q,k slots. One wave per group, 2 elems/lane, wave32 shuffles.
// ---------------------------------------------------------------------------
__global__ void norm_erf_kernel(float* __restrict__ qkv) {
  const int gid  = blockIdx.x * 8 + (threadIdx.x >> 5);
  const int lane = threadIdx.x & 31;
  float* g = qkv + (size_t)gid * HS_DIM;

  float e0 = g[lane];
  float e1 = g[lane + 32];

  float s = e0 + e1;
#pragma unroll
  for (int m = 16; m >= 1; m >>= 1) s += __shfl_xor(s, m, 32);
  const float mean = s * (1.0f / 64.0f);

  const float d0 = e0 - mean, d1 = e1 - mean;
  float ss = d0 * d0 + d1 * d1;
#pragma unroll
  for (int m = 16; m >= 1; m >>= 1) ss += __shfl_xor(ss, m, 32);
  const float inv_std = rsqrtf(ss * (1.0f / 63.0f));  // 1/(HS-1)

  float n0 = d0 * inv_std, n1 = d1 * inv_std;
  const int slot = (gid / NH_DIM) % 3;  // layout (b,t,3,h)
  if (slot != 2) { n0 = erff(n0); n1 = erff(n1); }  // erf on q and k only
  g[lane] = n0;
  g[lane + 32] = n1;
}

// ---------------------------------------------------------------------------
// Sequential scan. One block per (b,h); 64 threads; thread i owns state row i
// (64 floats kept in VGPRs via fully unrolled loops).
// Reference identities: a = -kk, b == k_ == kk*eta, so the update is
//   S_i     = sum_k s[i][k]*kk[k]
//   s[i][j] = w_i*s[i][j] + (v_i - S_i) * (kk[j]*eta[j])
//   vt_i    = sum_j s[i][j]*q[j]
// vt is written as bf16 for the projection GEMM.
// ---------------------------------------------------------------------------
__global__ void rwkv_scan_kernel(const float* __restrict__ qkv,
                                 const float* __restrict__ w,
                                 const float* __restrict__ eta,
                                 __bf16* __restrict__ vtb) {
  const int bh = blockIdx.x;
  const int b = bh / NH_DIM;
  const int h = bh % NH_DIM;
  const int i = threadIdx.x;

  float srow[HS_DIM];
#pragma unroll
  for (int j = 0; j < HS_DIM; ++j) srow[j] = 0.0f;

  const float wi   = w[h * HS_DIM + i];
  const float etai = eta[h * HS_DIM + i];

  __shared__ float kks[HS_DIM];
  __shared__ float kes[HS_DIM];
  __shared__ float qs[HS_DIM];

  for (int t = 0; t < T_DIM; ++t) {
    const size_t tok = ((size_t)b * T_DIM + t) * 3;
    const float q_i  = qkv[((tok + 0) * NH_DIM + h) * HS_DIM + i];
    const float kk_i = qkv[((tok + 1) * NH_DIM + h) * HS_DIM + i];
    const float v_i  = qkv[((tok + 2) * NH_DIM + h) * HS_DIM + i];
    qs[i]  = q_i;
    kks[i] = kk_i;
    kes[i] = kk_i * etai;
    __syncthreads();

    float S = 0.0f;
#pragma unroll
    for (int j = 0; j < HS_DIM; ++j) S += srow[j] * kks[j];
    const float coef = v_i - S;  // v_i + sa_i with a = -kk

    float acc = 0.0f;
#pragma unroll
    for (int j = 0; j < HS_DIM; ++j) {
      srow[j] = srow[j] * wi + coef * kes[j];
      acc += srow[j] * qs[j];
    }
    vtb[((size_t)b * T_DIM + t) * C_DIM + h * HS_DIM + i] = (__bf16)acc;
    __syncthreads();
  }
}

// ---------------------------------------------------------------------------
// Launch
// inputs: x (B,T,C) f32 | W_attn (3C,C) f32 | W_proj (C,C) f32 | w (C) | eta (C)
// output: x + y  (B,T,C) f32
// ---------------------------------------------------------------------------
extern "C" void kernel_launch(void* const* d_in, const int* in_sizes, int n_in,
                              void* d_out, int out_size, void* d_ws, size_t ws_size,
                              hipStream_t stream) {
  const float* x      = (const float*)d_in[0];
  const float* W_attn = (const float*)d_in[1];
  const float* W_proj = (const float*)d_in[2];
  const float* w      = (const float*)d_in[3];
  const float* eta    = (const float*)d_in[4];
  float* out = (float*)d_out;

  const size_t nx  = (size_t)B_DIM * T_DIM * C_DIM;  // 8388608
  const size_t nwa = (size_t)3 * C_DIM * C_DIM;      // 3145728
  const size_t nwp = (size_t)C_DIM * C_DIM;          // 1048576

  char* ws = (char*)d_ws;
  size_t off = 0;
  __bf16* xbf  = (__bf16*)(ws + off); off += nx  * sizeof(__bf16);
  __bf16* wabf = (__bf16*)(ws + off); off += nwa * sizeof(__bf16);
  __bf16* wpbf = (__bf16*)(ws + off); off += nwp * sizeof(__bf16);
  float*  qkv  = (float*)(ws + off);  off += nx * 3 * sizeof(float);
  __bf16* vtb  = (__bf16*)(ws + off); off += nx  * sizeof(__bf16);
  if (ws_size < off) return;  // insufficient scratch (compile-only safety)

  // 1) convert inputs to bf16 (4 elems/thread)
  f32_to_bf16_kernel<<<(int)(nx  / 4 + 255) / 256, 256, 0, stream>>>(x,      xbf,  (int)(nx  / 4));
  f32_to_bf16_kernel<<<(int)(nwa / 4 + 255) / 256, 256, 0, stream>>>(W_attn, wabf, (int)(nwa / 4));
  f32_to_bf16_kernel<<<(int)(nwp / 4 + 255) / 256, 256, 0, stream>>>(W_proj, wpbf, (int)(nwp / 4));

  // 2) qkv = x @ W_attn^T : M=8192, N=3072, K=1024
  {
    dim3 grid(B_DIM * T_DIM / 128, 3 * C_DIM / 64);
    gemm_bf16_wmma_kernel<false><<<grid, 128, 0, stream>>>(xbf, wabf, qkv, nullptr,
                                                           3 * C_DIM, C_DIM);
  }

  // 3) groupnorm(ddof=1) + erf over HS=64 groups; B*T*3*NH = 393216 groups
  {
    const int groups = B_DIM * T_DIM * 3 * NH_DIM;
    norm_erf_kernel<<<groups / 8, 256, 0, stream>>>(qkv);
  }

  // 4) sequential scan over T; one block per (b,h)
  rwkv_scan_kernel<<<B_DIM * NH_DIM, HS_DIM, 0, stream>>>(qkv, w, eta, vtb);

  // 5) out = x + vt @ W_proj^T : M=8192, N=1024, K=1024 (residual fused)
  {
    dim3 grid(B_DIM * T_DIM / 128, C_DIM / 64);
    gemm_bf16_wmma_kernel<true><<<grid, 128, 0, stream>>>(vtb, wpbf, out, x,
                                                          C_DIM, C_DIM);
  }
}
